// MultiheadedCausalSelfAttention_9216999817270
// MI455X (gfx1250) — compile-verified, matrix operand order checked by reading, not run on hardware
//
#include <hip/hip_runtime.h>

// Multi-head causal self-attention, MI455X (gfx1250, wave32, WMMA bf16, TDM async tile loads).
// Pipeline: cvt/transpose prep -> 3x QKV WMMA GEMM (TDM double-buffered) -> flash attention (WMMA)
//           -> output WMMA GEMM (TDM double-buffered).

#define B_  4
#define T_  2048
#define C_  1024
#define H_  16
#define D_  64
#define BT_ (B_ * T_)   // 8192

#define KSTEP 64
#define LSTR  72        // LDS row stride in bf16 (64 data + 8 pad) == TDM pad codes (4,3)

typedef __bf16 bf16;
typedef __bf16 v16bf __attribute__((ext_vector_type(16)));
typedef float  v8f   __attribute__((ext_vector_type(8)));
typedef unsigned int u32x4 __attribute__((ext_vector_type(4)));
typedef int i32x4 __attribute__((ext_vector_type(4)));
typedef int i32x8 __attribute__((ext_vector_type(8)));

union Frag16 { v16bf v; uint4 q[2]; };

// Gather one WMMA A/B fragment (16-bit, 16x32): elements 0..7 hold K=g*8.., 8..15 hold K=16+g*8..
__device__ __forceinline__ v16bf load_frag(const bf16* p, int g) {
  Frag16 f;
  f.q[0] = *(const uint4*)(p + g * 8);
  f.q[1] = *(const uint4*)(p + g * 8 + 16);
  return f.v;
}

// Wave-local LDS RAW ordering (LDS ops are in-order per wave; no block barrier needed).
__device__ __forceinline__ void wave_lds_sync() {
  asm volatile("s_wait_dscnt 0" ::: "memory");
}

#define WMMA(a, b, c) \
  __builtin_amdgcn_wmma_f32_16x16x32_bf16(false, (a), false, (b), (short)0, (c), false, false)

// ---- Tensor Data Mover: 2D bf16 tile -> LDS, with LDS padding for bank-conflict-free gathers.
// D# packing per CDNA5 ISA 8.3/8.4: group0 {count=1, lds_addr, global_addr, type=2},
// group1 {data_size=1(2B), pad_enable, pad_interval, pad_amount, tensor dims, tile dims, stride0}.
__device__ __forceinline__ void tdm_load_2d_bf16(const bf16* gp, unsigned ldsOff,
                                                 unsigned tensor0, unsigned tensor1,
                                                 unsigned stride0,
                                                 unsigned tile0, unsigned tile1,
                                                 unsigned padI, unsigned padA) {
  unsigned long long ga = (unsigned long long)(size_t)gp;
  u32x4 g0;
  g0[0] = 1u;                                            // count=1, user mode
  g0[1] = ldsOff;                                        // lds_addr (bytes)
  g0[2] = (unsigned)ga;                                  // global_addr[31:0]
  g0[3] = (unsigned)((ga >> 32) & 0x01FFFFFFu) | 0x80000000u;  // global_addr[56:32] | type=2
  i32x8 g1;
  g1[0] = (int)((1u << 16) | (1u << 20) | (padI << 22) | (padA << 25));
  g1[1] = (int)((tensor0 & 0xFFFFu) << 16);                                  // tensor_dim0 lo
  g1[2] = (int)(((tensor0 >> 16) & 0xFFFFu) | ((tensor1 & 0xFFFFu) << 16));  // dim0 hi | dim1 lo
  g1[3] = (int)(((tensor1 >> 16) & 0xFFFFu) | ((tile0 & 0xFFFFu) << 16));    // dim1 hi | tile0
  g1[4] = (int)(tile1 & 0xFFFFu);                                            // tile1 (tile2=0)
  g1[5] = (int)stride0;                                                      // stride0[31:0]
  g1[6] = 0;                                                                 // stride0 hi, stride1 lo
  g1[7] = 0;
  i32x4 z4 = {0, 0, 0, 0};
#if __clang_major__ >= 23
  i32x8 z8 = {0, 0, 0, 0, 0, 0, 0, 0};
  __builtin_amdgcn_tensor_load_to_lds(g0, g1, z4, z4, z8, 0);
#else
  __builtin_amdgcn_tensor_load_to_lds(g0, g1, z4, z4, 0);
#endif
}

// ---------------- prep kernels ----------------
__global__ __launch_bounds__(256) void cvt_bf16(const float* __restrict__ in,
                                                bf16* __restrict__ out, int n) {
  for (int i = blockIdx.x * 256 + threadIdx.x; i < n; i += gridDim.x * 256)
    out[i] = (bf16)in[i];
}

__global__ __launch_bounds__(256) void transpose_bf16(const float* __restrict__ W,
                                                      bf16* __restrict__ WT) {
  int i = blockIdx.x * 256 + threadIdx.x;      // over C_*C_ exactly
  int k = i >> 10, n = i & (C_ - 1);
  WT[(size_t)n * C_ + k] = (bf16)W[i];
}

// ---------------- QKV projection GEMM (TDM double-buffered) ----------------
// (BT x C) x (C x C): block tile 128x64, K-step 64, 8 waves each own a 16-row strip.
__global__ __launch_bounds__(256) void gemm_qkv(const bf16* __restrict__ X,
                                                const bf16* __restrict__ WT,
                                                const float* __restrict__ bias,
                                                bf16* __restrict__ out,
                                                int vTmode) {
  __shared__ bf16 sA[2][128 * LSTR];
  __shared__ bf16 sB[2][64 * LSTR];
  const int m0 = blockIdx.x * 128, n0 = blockIdx.y * 64;
  const int tid = threadIdx.x, w = tid >> 5, lane = tid & 31;
  const int l16 = lane & 15, g = lane >> 4;
  v8f acc[4] = {};
  const int NIT = C_ / KSTEP;   // 16
  if (tid < 32) {               // wave 0 drives the TDM pipeline
    tdm_load_2d_bf16(X + (size_t)m0 * C_, (unsigned)(size_t)&sA[0][0],
                     C_, BT_, C_, KSTEP, 128, 4, 3);
    tdm_load_2d_bf16(WT + (size_t)n0 * C_, (unsigned)(size_t)&sB[0][0],
                     C_, C_, C_, KSTEP, 64, 4, 3);
  }
  for (int it = 0; it < NIT; ++it) {
    const int p = it & 1;
    if (tid < 32) {
      if (it + 1 < NIT) {
        const int k0 = (it + 1) * KSTEP;
        tdm_load_2d_bf16(X + (size_t)m0 * C_ + k0, (unsigned)(size_t)&sA[p ^ 1][0],
                         C_, BT_, C_, KSTEP, 128, 4, 3);
        tdm_load_2d_bf16(WT + (size_t)n0 * C_ + k0, (unsigned)(size_t)&sB[p ^ 1][0],
                         C_, C_, C_, KSTEP, 64, 4, 3);
        __builtin_amdgcn_s_wait_tensorcnt(2);   // stage it's pair complete
      } else {
        __builtin_amdgcn_s_wait_tensorcnt(0);
      }
    }
    __syncthreads();
    #pragma unroll
    for (int kk = 0; kk < 2; ++kk) {
      v16bf a = load_frag(&sA[p][(w * 16 + l16) * LSTR + kk * 32], g);
      #pragma unroll
      for (int nt = 0; nt < 4; ++nt) {
        v16bf b = load_frag(&sB[p][(nt * 16 + l16) * LSTR + kk * 32], g);
        acc[nt] = WMMA(a, b, acc[nt]);
      }
    }
    __syncthreads();
  }
  #pragma unroll
  for (int nt = 0; nt < 4; ++nt)
    #pragma unroll
    for (int i = 0; i < 8; ++i) {
      int m = m0 + w * 16 + i + 8 * g;          // C/D layout: row = i + 8*(lane>=16)
      int n = n0 + nt * 16 + l16;
      float v = acc[nt][i] + bias[n];
      int bb = m >> 11, t = m & (T_ - 1);
      int h = n >> 6, d = n & (D_ - 1);
      size_t idx = vTmode ? (((size_t)(bb * H_ + h) * D_ + d) * T_ + t)
                          : (((size_t)(bb * H_ + h) * T_ + t) * D_ + d);
      out[idx] = (bf16)v;
    }
}

// ---------------- flash attention ----------------
// grid = (T/128, B*H); each wave owns 16 q rows, streams 32-key tiles to its causal bound.
__global__ __launch_bounds__(256) void attn_fwd(const bf16* __restrict__ Q,
                                                const bf16* __restrict__ K,
                                                const bf16* __restrict__ VT,
                                                bf16* __restrict__ Y) {
  __shared__ float sc[8][16 * 36];   // per-wave score tile (f32, padded)
  __shared__ bf16  pb[8][16 * 40];   // per-wave probs (bf16, padded for b128 frag gather)
  __shared__ float red[8][16];       // per-row alpha / l broadcast
  const int bh = blockIdx.y;
  const int tid = threadIdx.x, w = tid >> 5, lane = tid & 31;
  const int l16 = lane & 15, g = lane >> 4;
  const int qbase = blockIdx.x * 128 + w * 16;
  const bf16* qp = Q + (size_t)bh * T_ * D_;
  const bf16* kp = K + (size_t)bh * T_ * D_;
  const bf16* vp = VT + (size_t)bh * D_ * T_;
  v16bf qa[2];
  #pragma unroll
  for (int kk = 0; kk < 2; ++kk)
    qa[kk] = load_frag(qp + (size_t)(qbase + l16) * D_ + kk * 32, g);
  v8f o[4] = {};
  float m = -1e30f, l = 0.f;
  const float scale = 0.125f;        // 1/sqrt(64)
  const int nkt = ((qbase + 15) >> 5) + 1;
  for (int kt = 0; kt < nkt; ++kt) {
    if (kt + 1 < nkt) {              // prefetch next K/V tiles (global_prefetch)
      __builtin_prefetch(kp + (size_t)((kt + 1) * 32 + lane) * D_, 0, 0);
      __builtin_prefetch(vp + (size_t)lane * T_ + (kt + 1) * 32, 0, 0);
      __builtin_prefetch(vp + (size_t)(lane + 32) * T_ + (kt + 1) * 32, 0, 0);
    }
    // scores = q @ k^T : 2 N-tiles x 2 K-steps over D=64
    v8f s[2] = {};
    #pragma unroll
    for (int nt = 0; nt < 2; ++nt)
      #pragma unroll
      for (int kk = 0; kk < 2; ++kk) {
        v16bf kb = load_frag(kp + (size_t)(kt * 32 + nt * 16 + l16) * D_ + kk * 32, g);
        s[nt] = WMMA(qa[kk], kb, s[nt]);
      }
    // masked, scaled scores -> LDS (C layout scatter)
    #pragma unroll
    for (int nt = 0; nt < 2; ++nt)
      #pragma unroll
      for (int i = 0; i < 8; ++i) {
        int r = i + 8 * g, c = nt * 16 + l16;
        int key = kt * 32 + c;
        sc[w][r * 36 + c] = (key <= qbase + r) ? s[nt][i] * scale : -1e30f;
      }
    wave_lds_sync();
    // online softmax: lane pair (l16, l16+16) cooperates on row l16
    float pm = -1e30f;
    #pragma unroll
    for (int c = 0; c < 16; ++c)
      pm = fmaxf(pm, sc[w][l16 * 36 + g * 16 + c]);
    pm = fmaxf(pm, __shfl_xor(pm, 16, 32));
    float mn = fmaxf(m, pm);
    float alpha = __expf(m - mn);
    float ps = 0.f;
    #pragma unroll
    for (int c = 0; c < 16; ++c) {
      float e = __expf(sc[w][l16 * 36 + g * 16 + c] - mn);
      ps += e;
      pb[w][l16 * 40 + g * 16 + c] = (bf16)e;
    }
    ps += __shfl_xor(ps, 16, 32);
    l = alpha * l + ps;
    m = mn;
    if (g == 0) red[w][l16] = alpha;
    wave_lds_sync();
    // rescale running output by per-row alpha (broadcast through LDS)
    #pragma unroll
    for (int i = 0; i < 8; ++i) {
      float a8 = red[w][i + 8 * g];
      #pragma unroll
      for (int nt = 0; nt < 4; ++nt) o[nt][i] *= a8;
    }
    // o += p @ v : probs as A fragment from LDS, v^T rows give contiguous B fragments
    v16bf pa = load_frag(&pb[w][l16 * 40], g);
    #pragma unroll
    for (int nt = 0; nt < 4; ++nt) {
      v16bf vb = load_frag(vp + (size_t)(nt * 16 + l16) * T_ + kt * 32, g);
      o[nt] = WMMA(pa, vb, o[nt]);
    }
  }
  if (g == 0) red[w][l16] = l;
  wave_lds_sync();
  const int bb = bh >> 4, h = bh & (H_ - 1);
  #pragma unroll
  for (int i = 0; i < 8; ++i) {
    int r = i + 8 * g;
    float linv = 1.0f / red[w][r];
    int t = qbase + r;
    #pragma unroll
    for (int nt = 0; nt < 4; ++nt)
      Y[((size_t)bb * T_ + t) * C_ + h * D_ + nt * 16 + l16] = (bf16)(o[nt][i] * linv);
  }
}

// ---------------- output projection GEMM (TDM double-buffered) ----------------
__global__ __launch_bounds__(256) void gemm_out(const bf16* __restrict__ Yb,
                                                const bf16* __restrict__ WoT,
                                                const float* __restrict__ bo,
                                                float* __restrict__ out) {
  __shared__ bf16 sA[2][128 * LSTR];
  __shared__ bf16 sB[2][64 * LSTR];
  const int m0 = blockIdx.x * 128, n0 = blockIdx.y * 64;
  const int tid = threadIdx.x, w = tid >> 5, lane = tid & 31;
  const int l16 = lane & 15, g = lane >> 4;
  v8f acc[4] = {};
  const int NIT = C_ / KSTEP;
  if (tid < 32) {
    tdm_load_2d_bf16(Yb + (size_t)m0 * C_, (unsigned)(size_t)&sA[0][0],
                     C_, BT_, C_, KSTEP, 128, 4, 3);
    tdm_load_2d_bf16(WoT + (size_t)n0 * C_, (unsigned)(size_t)&sB[0][0],
                     C_, C_, C_, KSTEP, 64, 4, 3);
  }
  for (int it = 0; it < NIT; ++it) {
    const int p = it & 1;
    if (tid < 32) {
      if (it + 1 < NIT) {
        const int k0 = (it + 1) * KSTEP;
        tdm_load_2d_bf16(Yb + (size_t)m0 * C_ + k0, (unsigned)(size_t)&sA[p ^ 1][0],
                         C_, BT_, C_, KSTEP, 128, 4, 3);
        tdm_load_2d_bf16(WoT + (size_t)n0 * C_ + k0, (unsigned)(size_t)&sB[p ^ 1][0],
                         C_, C_, C_, KSTEP, 64, 4, 3);
        __builtin_amdgcn_s_wait_tensorcnt(2);
      } else {
        __builtin_amdgcn_s_wait_tensorcnt(0);
      }
    }
    __syncthreads();
    #pragma unroll
    for (int kk = 0; kk < 2; ++kk) {
      v16bf a = load_frag(&sA[p][(w * 16 + l16) * LSTR + kk * 32], g);
      #pragma unroll
      for (int nt = 0; nt < 4; ++nt) {
        v16bf b = load_frag(&sB[p][(nt * 16 + l16) * LSTR + kk * 32], g);
        acc[nt] = WMMA(a, b, acc[nt]);
      }
    }
    __syncthreads();
  }
  #pragma unroll
  for (int nt = 0; nt < 4; ++nt)
    #pragma unroll
    for (int i = 0; i < 8; ++i) {
      int mg = m0 + w * 16 + i + 8 * g;
      int ng = n0 + nt * 16 + l16;
      out[(size_t)mg * C_ + ng] = acc[nt][i] + bo[ng];
    }
}

// ---------------- launch ----------------
extern "C" void kernel_launch(void* const* d_in, const int* in_sizes, int n_in,
                              void* d_out, int out_size, void* d_ws, size_t ws_size,
                              hipStream_t stream) {
  (void)in_sizes; (void)n_in; (void)out_size; (void)ws_size;
  const float* x  = (const float*)d_in[0];
  const float* Wk = (const float*)d_in[1];
  const float* bk = (const float*)d_in[2];
  const float* Wq = (const float*)d_in[3];
  const float* bq = (const float*)d_in[4];
  const float* Wv = (const float*)d_in[5];
  const float* bv = (const float*)d_in[6];
  const float* Wo = (const float*)d_in[7];
  const float* bo = (const float*)d_in[8];
  float* out = (float*)d_out;
  char* ws = (char*)d_ws;
  const size_t MB = 1ull << 20;
  bf16* xb  = (bf16*)(ws + 0);        // 16 MB  x in bf16
  bf16* WqT = (bf16*)(ws + 16 * MB);  //  2 MB each: transposed bf16 weights
  bf16* WkT = (bf16*)(ws + 18 * MB);
  bf16* WvT = (bf16*)(ws + 20 * MB);
  bf16* WoT = (bf16*)(ws + 22 * MB);
  bf16* qb  = (bf16*)(ws + 24 * MB);  // 16 MB (B,H,T,D)
  bf16* kb  = (bf16*)(ws + 40 * MB);  // 16 MB (B,H,T,D)
  bf16* vTb = (bf16*)(ws + 56 * MB);  // 16 MB (B,H,D,T)
  bf16* yb  = (bf16*)(ws + 72 * MB);  // 16 MB (B,T,C)

  cvt_bf16<<<4096, 256, 0, stream>>>(x, xb, BT_ * C_);
  transpose_bf16<<<4096, 256, 0, stream>>>(Wq, WqT);
  transpose_bf16<<<4096, 256, 0, stream>>>(Wk, WkT);
  transpose_bf16<<<4096, 256, 0, stream>>>(Wv, WvT);
  transpose_bf16<<<4096, 256, 0, stream>>>(Wo, WoT);
  dim3 gg(64, 16);
  gemm_qkv<<<gg, 256, 0, stream>>>(xb, WqT, bq, qb, 0);
  gemm_qkv<<<gg, 256, 0, stream>>>(xb, WkT, bk, kb, 0);
  gemm_qkv<<<gg, 256, 0, stream>>>(xb, WvT, bv, vTb, 1);
  attn_fwd<<<dim3(16, 64), 256, 0, stream>>>(qb, kb, vTb, yb);
  gemm_out<<<gg, 256, 0, stream>>>(yb, WoT, bo, out);
}
